// MulticlassSequenceTaggerWithBahdanauAttention_37864431682586
// MI455X (gfx1250) — matrix operand order repaired and checked by reading, not compiled, not run on hardware
//
#include <hip/hip_runtime.h>
#include <hip/hip_bf16.h>

typedef __bf16 bf16_t;
typedef __attribute__((ext_vector_type(16))) __bf16 v16bf;
typedef __attribute__((ext_vector_type(8)))  __bf16 v8bf;
typedef __attribute__((ext_vector_type(8)))  float  v8f;

constexpr int B = 32, T = 512, E = 768, H = 512;
constexpr int IN = 2 * E;      // 1536 : LSTM input width [ctx, x_t]
constexpr int G4 = 4 * H;      // 2048 : gate width
constexpr int NWG = 64;        // persistent grid: one (dir,batch) per WG
constexpr int NTHREADS = 256;  // 8 wave32 per WG -> 512 waves total

// ---------------- workspace layout (bytes; every offset 256B aligned) ----------------
constexpr size_t OFF_CNT = 0;                                   // grid barrier counter
constexpr size_t OFF_C   = 256;                                 // c state  f32  [2][B][H]
constexpr size_t OFF_HBF = OFF_C   + (size_t)2*B*H*4;           // h state  bf16 [2][B][H]
constexpr size_t OFF_DEC = OFF_HBF + (size_t)2*B*H*2;           // dec      f32  [2][B][H]
constexpr size_t OFF_INP = OFF_DEC + (size_t)2*B*H*4;           // inp      bf16 [2][B][IN]
constexpr size_t OFF_GT  = OFF_INP + (size_t)2*B*IN*2;          // gates    f32  [2][B][G4]
constexpr size_t OFF_XBF = OFF_GT  + (size_t)2*B*G4*4;          // embeds   bf16 [B*T][E]
constexpr size_t OFF_EW  = OFF_XBF + (size_t)B*T*E*2;           // enc_W    bf16 [H][E]
constexpr size_t OFF_DW  = OFF_EW  + (size_t)H*E*2;             // dec_W    bf16 [H][H]
constexpr size_t OFF_WIH = OFF_DW  + (size_t)H*H*2;             // Wih      bf16 [2][G4][IN]
constexpr size_t OFF_WHH = OFF_WIH + (size_t)2*G4*IN*2;         // Whh      bf16 [2][G4][H]
constexpr size_t OFF_EP  = OFF_WHH + (size_t)2*G4*H*2;          // enc_proj f32  [B*T][H]
constexpr size_t OFF_HS  = OFF_EP  + (size_t)B*T*H*4;           // hs       bf16 [2][T][B][H]
constexpr size_t ZERO_BYTES = OFF_DEC;                          // counter + c + h

// ---------------- WMMA fragment helpers (ISA 7.12.2 layouts) ----------------
// A 16x32 bf16 (M x K), row-major source, ld in elements.
// lane&15 = M row; lane>>4 selects K-half: elems [8*kh,8*kh+8) and [16+8*kh, ...+8)
__device__ __forceinline__ v16bf frag_a(const bf16_t* base, int ld, int m0, int k0, int lane) {
    int m  = m0 + (lane & 15);
    int kh = ((lane >> 4) & 1) << 3;
    const bf16_t* p = base + (size_t)m * ld + k0 + kh;
    v8bf lo = *(const v8bf*)(p);
    v8bf hi = *(const v8bf*)(p + 16);
    return __builtin_shufflevector(lo, hi, 0,1,2,3,4,5,6,7,8,9,10,11,12,13,14,15);
}
// B 32x16 bf16 (K x N). Source is the row-major "N x K" weight matrix, so a row n
// supplies column n of B. lane>>4 selects K-half: elems [8*kh,8*kh+8) and [16+8*kh, ...+8)
__device__ __forceinline__ v16bf frag_b(const bf16_t* base, int ld, int n0, int k0, int lane) {
    int n = n0 + (lane & 15);
    int k = k0 + (((lane >> 4) & 1) << 3);
    const bf16_t* p = base + (size_t)n * ld + k;
    v8bf lo = *(const v8bf*)(p);
    v8bf hi = *(const v8bf*)(p + 16);
    return __builtin_shufflevector(lo, hi, 0,1,2,3,4,5,6,7,8,9,10,11,12,13,14,15);
}
__device__ __forceinline__ v8f wmma_bf16(v16bf a, v16bf b, v8f c) {
    return __builtin_amdgcn_wmma_f32_16x16x32_bf16(false, a, false, b, (short)0, c, false, false);
}
__device__ __forceinline__ float sigf(float x) { return 1.0f / (1.0f + __expf(-x)); }
__device__ __forceinline__ float wave_sum(float v) {
    #pragma unroll
    for (int m = 16; m > 0; m >>= 1) v += __shfl_xor(v, m, 32);
    return v;
}

// ---------------- f32 -> bf16 conversion ----------------
__global__ __launch_bounds__(256) void cvt_bf16_kernel(const float* __restrict__ src,
                                                       bf16_t* __restrict__ dst, size_t n) {
    size_t i = (size_t)blockIdx.x * blockDim.x + threadIdx.x;
    size_t s = (size_t)gridDim.x * blockDim.x;
    for (; i < n; i += s) dst[i] = (bf16_t)src[i];
}

// ---------------- enc_proj = embeds @ enc_W^T + enc_b (WMMA) ----------------
__global__ __launch_bounds__(256) void encproj_kernel(const bf16_t* __restrict__ Xbf,
                                                      const bf16_t* __restrict__ EWbf,
                                                      const float* __restrict__ enc_b,
                                                      float* __restrict__ EP) {
    const int lane   = threadIdx.x & 31;
    const int wave   = (blockIdx.x * blockDim.x + threadIdx.x) >> 5;
    const int nwaves = (gridDim.x * blockDim.x) >> 5;
    const int MT = (B * T) / 16, NT = H / 16;       // 1024 x 32 tiles
    for (int tile = wave; tile < MT * NT; tile += nwaves) {
        int mt = tile / NT, nt = tile % NT;
        v8f acc = {};
        #pragma unroll 4
        for (int k = 0; k < E; k += 32) {
            v16bf a = frag_a(Xbf,  E, mt * 16, k, lane);
            v16bf b = frag_b(EWbf, E, nt * 16, k, lane);
            acc = wmma_bf16(a, b, acc);
        }
        int n  = nt * 16 + (lane & 15);
        int mb = mt * 16 + ((lane >> 4) << 3);
        float bias = enc_b[n];
        #pragma unroll
        for (int r = 0; r < 8; ++r) EP[(size_t)(mb + r) * H + n] = acc[r] + bias;
    }
}

// ---------------- persistent recurrence kernel ----------------
__global__ __launch_bounds__(NTHREADS, 1) void recurrence_kernel(
        char* __restrict__ ws,
        const float* __restrict__ bih_f, const float* __restrict__ bhh_f,
        const float* __restrict__ bih_b, const float* __restrict__ bhh_b,
        const float* __restrict__ dec_b, const float* __restrict__ vvec) {

    unsigned* cnt      = (unsigned*)(ws + OFF_CNT);
    float*    c_state  = (float*)  (ws + OFF_C);
    bf16_t*   h_bf     = (bf16_t*) (ws + OFF_HBF);
    float*    dec      = (float*)  (ws + OFF_DEC);
    bf16_t*   inp      = (bf16_t*) (ws + OFF_INP);
    float*    gates    = (float*)  (ws + OFF_GT);
    const bf16_t* Xbf  = (const bf16_t*)(ws + OFF_XBF);
    const bf16_t* DW   = (const bf16_t*)(ws + OFF_DW);
    const bf16_t* WIH  = (const bf16_t*)(ws + OFF_WIH);
    const bf16_t* WHH  = (const bf16_t*)(ws + OFF_WHH);
    const float*  EP   = (const float*) (ws + OFF_EP);
    bf16_t*   hs       = (bf16_t*) (ws + OFF_HS);

    const int tid   = threadIdx.x;
    const int lane  = tid & 31;
    const int gwave = blockIdx.x * (NTHREADS >> 5) + (tid >> 5);   // 0..511
    const int gtid  = blockIdx.x * NTHREADS + tid;                 // 0..16383
    const int d_wg  = blockIdx.x >> 5;                             // direction for phase 2
    const int b_wg  = blockIdx.x & 31;                             // batch for phase 2

    __shared__ float s_dec[H];
    __shared__ float s_v[H];
    __shared__ float s_sc[T];
    __shared__ float s_red[NTHREADS];

    // v is loop-invariant: load once
    for (int i = tid; i < H; i += NTHREADS) s_v[i] = vvec[i];

    unsigned round = 0;
    auto gsync = [&]() {
        __syncthreads();
        if (tid == 0) {
            ++round;
            __threadfence();
            __hip_atomic_fetch_add(cnt, 1u, __ATOMIC_RELEASE, __HIP_MEMORY_SCOPE_AGENT);
            while (__hip_atomic_load(cnt, __ATOMIC_ACQUIRE, __HIP_MEMORY_SCOPE_AGENT) <
                   round * (unsigned)NWG)
                __builtin_amdgcn_s_sleep(2);
            __threadfence();
        } else {
            ++round;
        }
        __syncthreads();
    };
    __syncthreads();

    for (int step = 0; step < T; ++step) {
        // ---- phase 1: dec = h @ dec_W^T + dec_b  (2 x [32x512]x[512x512], WMMA)
        if (gwave < 128) {
            int d  = gwave >> 6;
            int r  = gwave & 63;
            int mt = r >> 5, nt = r & 31;
            const bf16_t* hA = h_bf + (size_t)d * B * H;
            v8f acc = {};
            #pragma unroll 4
            for (int k = 0; k < H; k += 32) {
                v16bf a = frag_a(hA, H, mt * 16, k, lane);
                v16bf b = frag_b(DW, H, nt * 16, k, lane);
                acc = wmma_bf16(a, b, acc);
            }
            int n  = nt * 16 + (lane & 15);
            int mb = mt * 16 + ((lane >> 4) << 3);
            float bias = dec_b[n];
            #pragma unroll
            for (int rr = 0; rr < 8; ++rr)
                dec[((size_t)d * B + (mb + rr)) * H + n] = acc[rr] + bias;
        }
        gsync();

        // ---- phase 2: attention scores -> softmax -> ctx ; build inp = [ctx, x_t]
        {
            const int d = d_wg, b = b_wg;
            const int tc = (d == 0) ? step : (T - 1 - step);
            for (int i = tid; i < H; i += NTHREADS)
                s_dec[i] = dec[((size_t)d * B + b) * H + i];
            __syncthreads();

            // scores[t] = sum_h v[h] * tanh(enc_proj[b,t,h] + dec[h])
            for (int t = (tid >> 5); t < T; t += (NTHREADS >> 5)) {
                const float* ep = EP + ((size_t)b * T + t) * H;
                float acc = 0.f;
                #pragma unroll 4
                for (int h = lane; h < H; h += 32)
                    acc += s_v[h] * tanhf(ep[h] + s_dec[h]);
                acc = wave_sum(acc);
                if (lane == 0) s_sc[t] = acc;
            }
            __syncthreads();

            // softmax over T=512 (two elements per thread)
            s_red[tid] = fmaxf(s_sc[tid], s_sc[tid + 256]);
            __syncthreads();
            for (int s = 128; s > 0; s >>= 1) {
                if (tid < s) s_red[tid] = fmaxf(s_red[tid], s_red[tid + s]);
                __syncthreads();
            }
            float mx = s_red[0];
            __syncthreads();
            float e0 = __expf(s_sc[tid] - mx), e1 = __expf(s_sc[tid + 256] - mx);
            s_sc[tid] = e0; s_sc[tid + 256] = e1;
            s_red[tid] = e0 + e1;
            __syncthreads();
            for (int s = 128; s > 0; s >>= 1) {
                if (tid < s) s_red[tid] += s_red[tid + s];
                __syncthreads();
            }
            float inv = 1.0f / s_red[0];
            __syncthreads();
            s_sc[tid] *= inv; s_sc[tid + 256] *= inv;
            __syncthreads();

            // ctx[e] = sum_t w[t] * embeds[b,t,e] ; inp = [ctx_bf16 | x_t_bf16]
            bf16_t* irow = inp + ((size_t)d * B + b) * IN;
            const bf16_t* xrow = Xbf + ((size_t)b * T + tc) * E;
            for (int e = tid; e < E; e += NTHREADS) {
                const bf16_t* xp = Xbf + (size_t)b * T * E + e;
                float acc = 0.f;
                #pragma unroll 4
                for (int t = 0; t < T; ++t)
                    acc += s_sc[t] * (float)xp[(size_t)t * E];
                irow[e]     = (bf16_t)acc;
                irow[E + e] = xrow[e];
            }
        }
        gsync();

        // ---- phase 3: gates = inp@Wih^T + h@Whh^T + bih + bhh  (WMMA, 1 tile/wave)
        {
            int d  = gwave >> 8;
            int r  = gwave & 255;
            int mt = r >> 7, nt = r & 127;
            const bf16_t* iA = inp  + (size_t)d * B * IN;
            const bf16_t* hA = h_bf + (size_t)d * B * H;
            const bf16_t* Wi = WIH  + (size_t)d * G4 * IN;
            const bf16_t* Wh = WHH  + (size_t)d * G4 * H;
            const float* bih = d ? bih_b : bih_f;
            const float* bhh = d ? bhh_b : bhh_f;
            v8f acc = {};
            #pragma unroll 4
            for (int k = 0; k < IN; k += 32) {
                v16bf a = frag_a(iA, IN, mt * 16, k, lane);
                v16bf b = frag_b(Wi, IN, nt * 16, k, lane);
                acc = wmma_bf16(a, b, acc);
            }
            #pragma unroll 4
            for (int k = 0; k < H; k += 32) {
                v16bf a = frag_a(hA, H, mt * 16, k, lane);
                v16bf b = frag_b(Wh, H, nt * 16, k, lane);
                acc = wmma_bf16(a, b, acc);
            }
            int n  = nt * 16 + (lane & 15);
            int mb = mt * 16 + ((lane >> 4) << 3);
            float bias = bih[n] + bhh[n];
            #pragma unroll
            for (int rr = 0; rr < 8; ++rr)
                gates[((size_t)d * B + (mb + rr)) * G4 + n] = acc[rr] + bias;
        }
        gsync();

        // ---- phase 4: LSTM pointwise; update c (f32), h (bf16), emit hs[t]
        for (int u = gtid; u < 2 * B * H; u += NWG * NTHREADS) {
            int d = u / (B * H);
            int rem = u - d * (B * H);
            int b = rem / H, h = rem - (rem / H) * H;
            const float* g = gates + ((size_t)d * B + b) * G4;
            float gi = g[h], gf = g[H + h], gg = g[2 * H + h], go = g[3 * H + h];
            float c2 = sigf(gf) * c_state[u] + sigf(gi) * tanhf(gg);
            float h2 = sigf(go) * tanhf(c2);
            c_state[u] = c2;
            h_bf[u]    = (bf16_t)h2;
            int tc = (d == 0) ? step : (T - 1 - step);
            hs[(((size_t)d * T + tc) * B + b) * H + h] = (bf16_t)h2;
        }
        gsync();
    }
}

// ---------------- classifier head: out[b,t,0:5]=para, [5:8]=chap ----------------
__global__ __launch_bounds__(256) void head_kernel(const bf16_t* __restrict__ hs,
                                                   const float* __restrict__ paraW,
                                                   const float* __restrict__ parab,
                                                   const float* __restrict__ chapW,
                                                   const float* __restrict__ chapb,
                                                   float* __restrict__ out) {
    const int lane   = threadIdx.x & 31;
    const int wave   = (blockIdx.x * blockDim.x + threadIdx.x) >> 5;
    const int nwaves = (gridDim.x * blockDim.x) >> 5;
    for (int bt = wave; bt < B * T; bt += nwaves) {
        int b = bt / T, t = bt - (bt / T) * T;
        const bf16_t* hf = hs + ((size_t)t * B + b) * H;               // fwd
        const bf16_t* hb = hs + (((size_t)T + t) * B + b) * H;         // bwd
        #pragma unroll
        for (int c = 0; c < 8; ++c) {
            const float* W = (c < 5) ? (paraW + (size_t)c * 2 * H)
                                     : (chapW + (size_t)(c - 5) * 2 * H);
            float acc = 0.f;
            for (int k = lane; k < H; k += 32)
                acc += W[k] * (float)hf[k] + W[H + k] * (float)hb[k];
            acc = wave_sum(acc);
            if (lane == 0)
                out[(size_t)(b * T + t) * 8 + c] = acc + ((c < 5) ? parab[c] : chapb[c - 5]);
        }
    }
}

extern "C" void kernel_launch(void* const* d_in, const int* in_sizes, int n_in,
                              void* d_out, int out_size, void* d_ws, size_t ws_size,
                              hipStream_t stream) {
    const float* embeds = (const float*)d_in[0];
    const float* Wih_f  = (const float*)d_in[1];
    const float* Whh_f  = (const float*)d_in[2];
    const float* bih_f  = (const float*)d_in[3];
    const float* bhh_f  = (const float*)d_in[4];
    const float* Wih_b  = (const float*)d_in[5];
    const float* Whh_b  = (const float*)d_in[6];
    const float* bih_b  = (const float*)d_in[7];
    const float* bhh_b  = (const float*)d_in[8];
    const float* enc_W  = (const float*)d_in[9];
    const float* enc_b  = (const float*)d_in[10];
    const float* dec_W  = (const float*)d_in[11];
    const float* dec_b  = (const float*)d_in[12];
    const float* vvec   = (const float*)d_in[13];
    const float* paraW  = (const float*)d_in[14];
    const float* parab  = (const float*)d_in[15];
    const float* chapW  = (const float*)d_in[16];
    const float* chapb  = (const float*)d_in[17];
    float* out = (float*)d_out;
    char*  ws  = (char*)d_ws;

    // zero barrier counter + (h, c) initial state
    hipMemsetAsync(ws, 0, ZERO_BYTES, stream);

    // f32 -> bf16 operand staging
    cvt_bf16_kernel<<<2048, 256, 0, stream>>>(embeds, (bf16_t*)(ws + OFF_XBF), (size_t)B*T*E);
    cvt_bf16_kernel<<<256,  256, 0, stream>>>(enc_W,  (bf16_t*)(ws + OFF_EW),  (size_t)H*E);
    cvt_bf16_kernel<<<256,  256, 0, stream>>>(dec_W,  (bf16_t*)(ws + OFF_DW),  (size_t)H*H);
    cvt_bf16_kernel<<<1024, 256, 0, stream>>>(Wih_f,  (bf16_t*)(ws + OFF_WIH), (size_t)G4*IN);
    cvt_bf16_kernel<<<1024, 256, 0, stream>>>(Wih_b,  (bf16_t*)(ws + OFF_WIH) + (size_t)G4*IN, (size_t)G4*IN);
    cvt_bf16_kernel<<<512,  256, 0, stream>>>(Whh_f,  (bf16_t*)(ws + OFF_WHH), (size_t)G4*H);
    cvt_bf16_kernel<<<512,  256, 0, stream>>>(Whh_b,  (bf16_t*)(ws + OFF_WHH) + (size_t)G4*H, (size_t)G4*H);

    // enc_proj precompute (WMMA)
    encproj_kernel<<<256, 256, 0, stream>>>((const bf16_t*)(ws + OFF_XBF),
                                            (const bf16_t*)(ws + OFF_EW),
                                            enc_b, (float*)(ws + OFF_EP));

    // persistent bidirectional attention-LSTM recurrence
    recurrence_kernel<<<NWG, NTHREADS, 0, stream>>>(ws, bih_f, bhh_f, bih_b, bhh_b,
                                                    dec_b, vvec);

    // classifier head
    head_kernel<<<128, 256, 0, stream>>>((const bf16_t*)(ws + OFF_HS),
                                         paraW, parab, chapW, chapb, out);
    (void)in_sizes; (void)n_in; (void)out_size; (void)ws_size;
}
